// WindowAttention_21947282882638
// MI455X (gfx1250) — compile-verified, matrix operand order checked by reading, not run on hardware
//
#include <hip/hip_runtime.h>

typedef __attribute__((ext_vector_type(16))) _Float16 v16h;
typedef __attribute__((ext_vector_type(8)))  float    v8f;
typedef __attribute__((ext_vector_type(4)))  unsigned int u32x4;
typedef __attribute__((ext_vector_type(8)))  int      i32x8;
typedef __attribute__((ext_vector_type(4)))  int      i32x4;

#define C_DIM   256
#define HW      128

#define HAS_TDM __has_builtin(__builtin_amdgcn_tensor_load_to_lds)

// LDS layout (bytes). Regions are reused across phases:
//  OFF_X : f32 [256][64] raw input, channel-major TDM tile (65536B);
//          later f16 probs [8][64][72] (73728B)
//  OFF_A : f16 [64][264] LN output; later attention output (A of out-proj)
//  OFF_Q : f16 [64][264] Q;  OFF_K: f16 [64][264] K
//  OFF_VT: f16 [256][72] V transposed (row = channel, col = token)
//  OFF_OUT (= OFF_Q): f32 [256][64] out-proj result, channel-major (aliases dead Q+K)
#define OFF_X   0
#define OFF_A   73728
#define OFF_Q   107520
#define OFF_K   141312
#define OFF_VT  175104
#define SMEM_BYTES 211968
#define OFF_OUT OFF_Q

// Load a 16x32 f16 WMMA fragment (A layout; B uses the same per-lane pattern
// when the source is stored as Wrow[n][k] row-major). Per ISA 7.12.2:
// lanes 0-15: row=lane, halves 0-7 = K kb..kb+7, halves 8-15 = K kb+16..kb+23
// lanes 16-31: same rows, kb += 8.
__device__ __forceinline__ v16h load_frag(const _Float16* base, int row0, int k0, int stride) {
  int lane = threadIdx.x & 31;
  const _Float16* p = base + (row0 + (lane & 15)) * stride + k0 + ((lane >> 4) << 3);
  union { v16h v; u32x4 u[2]; } x;
  x.u[0] = *(const u32x4*)(p);
  x.u[1] = *(const u32x4*)(p + 16);
  return x.v;
}

__device__ __forceinline__ v8f wmma_f16(v16h a, v16h b, v8f c) {
  return __builtin_amdgcn_wmma_f32_16x16x32_f16(false, a, false, b, (short)0, c, false, false);
}

#if HAS_TDM
// D# for an 8x8x256 tile of a [C=256][H=128][W=128] f32 tensor (strides 128 / 16384).
__device__ __forceinline__ void tdm_desc(u32x4& g0, i32x8& g1,
                                         unsigned lds_byte_off, unsigned long long gaddr) {
  g0 = (u32x4){1u,                                   // count=1, is_restore=0
               lds_byte_off,                         // lds_addr
               (unsigned)gaddr,                      // global_addr[31:0]
               ((unsigned)(gaddr >> 32) & 0x01FFFFFFu) | 0x80000000u}; // addr[56:32] | type=2
  g1 = (i32x8){(int)(2u << 16),                      // workgroup_mask=0, data_size=2 (4B)
               (int)(128u << 16),                    // tensor_dim0=128 (low16 @ [31:16])
               (int)(128u << 16),                    // tensor_dim0 hi=0 | tensor_dim1 lo16=128
               (int)(8u << 16),                      // tensor_dim1 hi=0 | tile_dim0=8
               (int)(8u | (256u << 16)),             // tile_dim1=8 | tile_dim2=256
               (int)128,                             // tensor_dim0_stride lo32 = 128
               (int)(0x4000u << 16),                 // stride0 hi=0 | tensor_dim1_stride lo16=16384
               0};                                   // tensor_dim1_stride hi = 0
}
#endif

extern "C" __global__ void convert_weights_kernel(const float* __restrict__ w_in,
                                                  const float* __restrict__ w_out,
                                                  _Float16* __restrict__ w16) {
  int i = blockIdx.x * 256 + threadIdx.x;
  if (i < 768 * 256) w16[i] = (_Float16)w_in[i];
  if (i < 256 * 256) w16[768 * 256 + i] = (_Float16)w_out[i];
}

extern "C" __global__ __launch_bounds__(256, 1)
void win_attn_kernel(const float* __restrict__ x,
                     const float* __restrict__ ln_g, const float* __restrict__ ln_b,
                     const float* __restrict__ b_in, const float* __restrict__ b_out,
                     const _Float16* __restrict__ w16_in, const _Float16* __restrict__ w16_out,
                     float* __restrict__ out)
{
  extern __shared__ char smem[];
  float*    s_x  = (float*)(smem + OFF_X);     // [256][64] channel-major
  _Float16* s_a  = (_Float16*)(smem + OFF_A);
  _Float16* s_q  = (_Float16*)(smem + OFF_Q);
  _Float16* s_k  = (_Float16*)(smem + OFF_K);
  _Float16* s_vt = (_Float16*)(smem + OFF_VT);
  _Float16* s_pr = (_Float16*)(smem + OFF_X);
  float*    s_o  = (float*)(smem + OFF_OUT);   // [256][64] channel-major

  const int tid  = threadIdx.x;
  const int lane = tid & 31;
  const int wave = tid >> 5;
  const int win  = blockIdx.x;                 // 4096 windows
  const int b    = win >> 8;
  const int wy   = (win >> 4) & 15;
  const int wx   = win & 15;
  const size_t plane = (size_t)HW * HW;        // 16384
  const float* xb = x   + (size_t)b * C_DIM * plane;
  float*       ob = out + (size_t)b * C_DIM * plane;
  const int h0 = wy * 8, w0 = wx * 8;

  // ---------------- gather window tile into LDS (TDM) ----------------
#if HAS_TDM
  if (wave == 0) {
    u32x4 g0; i32x8 g1;
    i32x4 g2 = (i32x4){256, 0, 0, 0};          // tensor_dim2 = 256
    i32x4 g3 = (i32x4){0, 0, 0, 0};
    i32x8 g4 = (i32x8){0, 0, 0, 0, 0, 0, 0, 0};
    unsigned long long ga = (unsigned long long)(uintptr_t)xb
                          + (unsigned long long)(h0 * HW + w0) * 4ull;
    tdm_desc(g0, g1, (unsigned)(uintptr_t)(void*)s_x, ga);
    __builtin_amdgcn_tensor_load_to_lds(g0, g1, g2, g3, g4, 0);
    __builtin_amdgcn_s_wait_tensorcnt(0);
  }
#else
  {
    int token = tid & 63, csub = tid >> 6;
    size_t gb = (size_t)(h0 + (token >> 3)) * HW + (w0 + (token & 7));
    #pragma unroll 4
    for (int co = 0; co < 64; ++co) {
      int c = co * 4 + csub;
      s_x[c * 64 + token] = xb[(size_t)c * plane + gb];
    }
  }
#endif
  __syncthreads();

  // ---------------- LayerNorm -> f16 A matrix ----------------
  {
    int tok = tid >> 2, part = tid & 3;
    int c0 = part * 64;
    float s = 0.f, ss = 0.f;
    #pragma unroll 8
    for (int i = 0; i < 64; ++i) { float v = s_x[(c0 + i) * 64 + tok]; s += v; ss += v * v; }
    s  += __shfl_xor(s, 1, 32);  s  += __shfl_xor(s, 2, 32);
    ss += __shfl_xor(ss, 1, 32); ss += __shfl_xor(ss, 2, 32);
    float mu = s * (1.f / 256.f);
    float rstd = rsqrtf(ss * (1.f / 256.f) - mu * mu + 1e-5f);
    #pragma unroll 8
    for (int i = 0; i < 64; ++i) {
      int c = c0 + i;
      float v = (s_x[c * 64 + tok] - mu) * rstd * ln_g[c] + ln_b[c];
      s_a[tok * 264 + c] = (_Float16)v;
    }
  }
  __syncthreads();

  // ---------------- QKV GEMM: (64x256) x (256x768), WMMA f16 ----------------
  {
    const float qscale = 0.17677669529663687f;   // 1/sqrt(32)
    for (int i = 0; i < 4; ++i) {
      v16h afr[8];
      #pragma unroll
      for (int kk = 0; kk < 8; ++kk) afr[kk] = load_frag(s_a, i * 16, kk * 32, 264);
      for (int jj = 0; jj < 6; ++jj) {
        int j = wave + jj * 8;                   // 48 N-tiles split over 8 waves
        int n0 = j * 16;
        v8f acc = {};
        #pragma unroll
        for (int kk = 0; kk < 8; ++kk) {
          v16h bfr = load_frag(w16_in, n0, kk * 32, 256);   // w_in row-major = B[k][n]
          acc = wmma_f16(afr[kk], bfr, acc);
        }
        int colg = n0 + (lane & 15);
        float bias = b_in[colg];
        int m0 = i * 16 + (lane >> 4) * 8;
        if (n0 < 256) {            // Q (pre-scaled), row-major
          #pragma unroll
          for (int r = 0; r < 8; ++r)
            s_q[(m0 + r) * 264 + colg] = (_Float16)((acc[r] + bias) * qscale);
        } else if (n0 < 512) {     // K, row-major
          int c = colg - 256;
          #pragma unroll
          for (int r = 0; r < 8; ++r)
            s_k[(m0 + r) * 264 + c] = (_Float16)(acc[r] + bias);
        } else {                   // V, transposed: lane holds 8 consecutive rows -> b128 store
          int c = colg - 512;
          union { u32x4 u; _Float16 hh[8]; } t;
          #pragma unroll
          for (int r = 0; r < 8; ++r) t.hh[r] = (_Float16)(acc[r] + bias);
          *(u32x4*)(&s_vt[c * 72 + m0]) = t.u;
        }
      }
    }
  }
  __syncthreads();

  // ---------------- attention: one head per wave ----------------
  {
    const int h = wave;
    _Float16* pr = s_pr + h * (64 * 72);
    v8f acc[4][4];
    #pragma unroll
    for (int i = 0; i < 4; ++i) {
      v16h aq = load_frag(s_q, i * 16, h * 32, 264);     // K=dh=32 -> one k-step
      #pragma unroll
      for (int j = 0; j < 4; ++j) {
        v16h bk = load_frag(s_k, j * 16, h * 32, 264);   // k[n][d] row-major = B[d][n]
        v8f z = {};
        acc[i][j] = wmma_f16(aq, bk, z);
      }
    }
    // softmax over 64 columns: reduce across 4 tiles + 16 lanes of each half
    #pragma unroll
    for (int i = 0; i < 4; ++i) {
      #pragma unroll
      for (int r = 0; r < 8; ++r) {
        float m = fmaxf(fmaxf(acc[i][0][r], acc[i][1][r]),
                        fmaxf(acc[i][2][r], acc[i][3][r]));
        m = fmaxf(m, __shfl_xor(m, 1, 32));
        m = fmaxf(m, __shfl_xor(m, 2, 32));
        m = fmaxf(m, __shfl_xor(m, 4, 32));
        m = fmaxf(m, __shfl_xor(m, 8, 32));
        float s = 0.f;
        #pragma unroll
        for (int j = 0; j < 4; ++j) {
          float e = __expf(acc[i][j][r] - m);
          acc[i][j][r] = e;
          s += e;
        }
        s += __shfl_xor(s, 1, 32);
        s += __shfl_xor(s, 2, 32);
        s += __shfl_xor(s, 4, 32);
        s += __shfl_xor(s, 8, 32);
        float rs = 1.f / s;
        #pragma unroll
        for (int j = 0; j < 4; ++j) acc[i][j][r] *= rs;
      }
      int m0 = i * 16 + (lane >> 4) * 8;
      int colb = lane & 15;
      #pragma unroll
      for (int j = 0; j < 4; ++j) {
        int col = j * 16 + colb;
        #pragma unroll
        for (int r = 0; r < 8; ++r)
          pr[(m0 + r) * 72 + col] = (_Float16)acc[i][j][r];
      }
    }
    // out = attn @ v : A = probs [64][72], B = s_vt (v already transposed)
    #pragma unroll
    for (int i = 0; i < 4; ++i) {
      #pragma unroll
      for (int jn = 0; jn < 2; ++jn) {
        v8f o = {};
        #pragma unroll
        for (int kk = 0; kk < 2; ++kk) {
          v16h ap = load_frag(pr, i * 16, kk * 32, 72);
          v16h bv = load_frag(s_vt, h * 32 + jn * 16, kk * 32, 72);
          o = wmma_f16(ap, bv, o);
        }
        int col = h * 32 + jn * 16 + (lane & 15);
        int m0 = i * 16 + (lane >> 4) * 8;
        #pragma unroll
        for (int r = 0; r < 8; ++r)
          s_a[(m0 + r) * 264 + col] = (_Float16)o[r];
      }
    }
  }
  __syncthreads();

  // ---------------- output projection: (64x256) x (256x256) ----------------
  {
    for (int i = 0; i < 4; ++i) {
      v16h afr[8];
      #pragma unroll
      for (int kk = 0; kk < 8; ++kk) afr[kk] = load_frag(s_a, i * 16, kk * 32, 264);
      #pragma unroll
      for (int jj = 0; jj < 2; ++jj) {
        int n0 = (wave * 2 + jj) * 16;
        v8f acc = {};
        #pragma unroll
        for (int kk = 0; kk < 8; ++kk) {
          v16h bfr = load_frag(w16_out, n0, kk * 32, 256);
          acc = wmma_f16(afr[kk], bfr, acc);
        }
        int col = n0 + (lane & 15);            // output channel
        float bias = b_out[col];
        int m0 = i * 16 + (lane >> 4) * 8;
        // channel-major staging: lane holds 8 consecutive tokens of one channel
        union { u32x4 u[2]; float f[8]; } t;
        #pragma unroll
        for (int r = 0; r < 8; ++r) t.f[r] = acc[r] + bias;
        *(u32x4*)(&s_o[col * 64 + m0])     = t.u[0];
        *(u32x4*)(&s_o[col * 64 + m0 + 4]) = t.u[1];
      }
    }
  }
  __syncthreads();

  // ---------------- scatter tile back to (B,C,H,W) (TDM) ----------------
#if HAS_TDM
  if (wave == 0) {
    u32x4 g0; i32x8 g1;
    i32x4 g2 = (i32x4){256, 0, 0, 0};
    i32x4 g3 = (i32x4){0, 0, 0, 0};
    i32x8 g4 = (i32x8){0, 0, 0, 0, 0, 0, 0, 0};
    unsigned long long ga = (unsigned long long)(uintptr_t)ob
                          + (unsigned long long)(h0 * HW + w0) * 4ull;
    tdm_desc(g0, g1, (unsigned)(uintptr_t)(void*)s_o, ga);
    __builtin_amdgcn_tensor_store_from_lds(g0, g1, g2, g3, g4, 0);
    __builtin_amdgcn_s_wait_tensorcnt(0);
  }
#else
  {
    int token = tid & 63, csub = tid >> 6;
    size_t gb = (size_t)(h0 + (token >> 3)) * HW + (w0 + (token & 7));
    #pragma unroll 4
    for (int co = 0; co < 64; ++co) {
      int c = co * 4 + csub;
      ob[(size_t)c * plane + gb] = s_o[c * 64 + token];
    }
  }
#endif
}

extern "C" void kernel_launch(void* const* d_in, const int* in_sizes, int n_in,
                              void* d_out, int out_size, void* d_ws, size_t ws_size,
                              hipStream_t stream) {
  // setup_inputs order: x, ln_gamma, ln_beta, w_in, b_in, w_out, b_out (all f32)
  const float* x     = (const float*)d_in[0];
  const float* ln_g  = (const float*)d_in[1];
  const float* ln_b  = (const float*)d_in[2];
  const float* w_in  = (const float*)d_in[3];
  const float* b_in  = (const float*)d_in[4];
  const float* w_out = (const float*)d_in[5];
  const float* b_out = (const float*)d_in[6];

  _Float16* w16 = (_Float16*)d_ws;             // 768*256 + 256*256 halves = 512 KB

  convert_weights_kernel<<<768, 256, 0, stream>>>(w_in, w_out, w16);

  win_attn_kernel<<<4096, 256, SMEM_BYTES, stream>>>(
      x, ln_g, ln_b, b_in, b_out, w16, w16 + 768 * 256, (float*)d_out);
}
